// VQVae_31129922961758
// MI455X (gfx1250) — matrix operand order repaired and checked by reading, not compiled, not run on hardware
//
#include <hip/hip_runtime.h>

// ---------------------------------------------------------------------------
// CDNA5 (gfx1250) VQ-VAE forward.  All conv1d layers + the codebook GEMM run
// as implicit GEMMs on v_wmma_f32_16x16x32_f16 (wave32, f32 accumulate).
//  - weights pre-packed to f16 fragment layout, zero padded to 128 M-rows
//  - activations f16 with zeroed time halos -> branch-free inner loop
//  - B operands staged in LDS transposed, read as aligned ds_load_b128
//  - wave computes 16x64 (4 N-tiles): A reused across 4 WMMAs; B fragments
//    preloaded per tap so the compiler can emit graduated dscnt waits
// ---------------------------------------------------------------------------

typedef __attribute__((ext_vector_type(16))) _Float16 v16h;
typedef __attribute__((ext_vector_type(8)))  _Float16 v8h;
typedef __attribute__((ext_vector_type(8)))  float    v8f;

union CF { v8f v; float f[8]; };
union AB { v16h v; v8h q[2]; };

#define NFEATS 63
#define WIDTH  512
#define CDIM   512     // channel capacity of activation buffers
#define CODEN  512
#define CODED  512
#define BATCH  32
#define TFULL  1024
#define TQ     128
#define HALO   16
#define TSTRIDE (TFULL + 2 * HALO)   // 1056, fixed channel stride (halves)
#define LROW   40                    // LDS row stride in halves (80B, 16B-aligned b128)

// ---------------------------------------------------------------------------
// Implicit-GEMM conv1d.  Wave tile = 16(Cout) x 64(T); block = 8 waves stacked
// over M = 128(Cout) x 64(T).  grid = (Tout/64, Coutp128/128, B).
// ---------------------------------------------------------------------------
template <int K, int STRIDE, int DIL>
__global__ __launch_bounds__(256) void conv1d_wmma(
    const _Float16* __restrict__ x,   // halo layout (B, CDIM, TSTRIDE), data at +HALO
    const _Float16* __restrict__ wh,  // packed (Coutp128, K, Cinp) f16, zero padded
    const float*    __restrict__ bias,// (Cout) or null
    const _Float16* __restrict__ res, // halo layout or null
    _Float16*       __restrict__ yh,  // halo layout or null
    float*          __restrict__ yf,  // compact (B, Cout, Tout) or null
    int Cinp, int Cout, int Tout, int pad, int relu_in, int relu_out)
{
  constexpr int WLEN = 63 * STRIDE + (K - 1) * DIL + 1;   // staged time window
  __shared__ _Float16 lds[WLEN * LROW];

  const int lane = threadIdx.x & 31;
  const int wid  = threadIdx.x >> 5;
  const int b    = blockIdx.z;
  const int m0   = (blockIdx.y * 8 + wid) * 16;
  const int nblk = blockIdx.x * 64;
  const int lm   = lane & 15;
  const int hi   = lane >> 4;
  const int tmin = nblk * STRIDE - pad;

  const size_t xb = (size_t)b * CDIM * TSTRIDE;
  const int m = m0 + lm;                 // A row, always < Coutp128

  CF acc[4];
  #pragma unroll
  for (int nt = 0; nt < 4; ++nt)
    #pragma unroll
    for (int r = 0; r < 8; ++r) acc[nt].f[r] = 0.f;

  const _Float16* wbase = wh + ((size_t)m * K) * Cinp + (hi << 3);

  for (int cc = 0; cc < Cinp; cc += 32) {
    // ---- stage x[cc..cc+32) x [tmin, tmin+WLEN) into LDS transposed [t][ci]
    #pragma unroll
    for (int cj = 0; cj < 4; ++cj) {
      const int cr = wid + cj * 8;
      const _Float16* src = x + xb + (size_t)(cc + cr) * TSTRIDE + (HALO + tmin);
      #pragma unroll
      for (int tt = 0; tt < (WLEN + 31) / 32; ++tt) {
        int t = lane + tt * 32;
        if ((WLEN & 31) == 0 || t < WLEN) {
          float v = (float)src[t];
          if (relu_in) v = fmaxf(v, 0.f);
          lds[t * LROW + cr] = (_Float16)v;
        }
      }
    }
    __syncthreads();

    // ---- A fragments for all taps: one clause of aligned b128 loads
    AB a[K];
    #pragma unroll
    for (int tap = 0; tap < K; ++tap) {
      a[tap].q[0] = *(const v8h*)(wbase + cc + (size_t)tap * Cinp);
      a[tap].q[1] = *(const v8h*)(wbase + cc + (size_t)tap * Cinp + 16);
    }
    // ---- WMMA burst: per tap preload all 4 B fragments (8 ds_load_b128),
    //      then 4 WMMAs -> graduated dscnt waits instead of full stalls
    #pragma unroll
    for (int tap = 0; tap < K; ++tap) {
      AB bb[4];
      #pragma unroll
      for (int nt = 0; nt < 4; ++nt) {
        const _Float16* lsrc =
            lds + ((nt * 16 + lm) * STRIDE + tap * DIL) * LROW + (hi << 3);
        bb[nt].q[0] = *(const v8h*)(lsrc);
        bb[nt].q[1] = *(const v8h*)(lsrc + 16);
      }
      #pragma unroll
      for (int nt = 0; nt < 4; ++nt) {
        acc[nt].v = __builtin_amdgcn_wmma_f32_16x16x32_f16(
            false, a[tap].v, false, bb[nt].v, (short)0, acc[nt].v, false, false);
      }
    }
    __syncthreads();
  }

  // ---- epilogue: bias + residual + relu, store f16 halo and/or f32 compact
  #pragma unroll
  for (int nt = 0; nt < 4; ++nt) {
    #pragma unroll
    for (int r = 0; r < 8; ++r) {
      int mm = m0 + r + (hi << 3);
      int nn = nblk + nt * 16 + lm;
      if (mm < Cout) {
        float v = acc[nt].f[r];
        if (bias) v += bias[mm];
        if (res) v += (float)res[xb + (size_t)mm * TSTRIDE + HALO + nn];
        if (relu_out) v = fmaxf(v, 0.f);
        if (yh) yh[xb + (size_t)mm * TSTRIDE + HALO + nn] = (_Float16)v;
        if (yf) yf[((size_t)b * Cout + mm) * Tout + nn] = v;
      }
    }
  }
}

// ---------------------------------------------------------------------------
// Pack conv weights (Cout, Cin, K) f32 -> (Coutp128, K, Cinp) f16, zero padded.
// ---------------------------------------------------------------------------
__global__ void pack_w(const float* __restrict__ w, _Float16* __restrict__ wh,
                       int Cout, int Cin, int K, int Cinp, int total)
{
  int id = blockIdx.x * blockDim.x + threadIdx.x;
  if (id >= total) return;
  int ci  = id % Cinp;
  int tap = (id / Cinp) % K;
  int m   = id / (Cinp * K);
  float v = 0.f;
  if (m < Cout && ci < Cin) v = w[((size_t)m * Cin + ci) * K + tap];
  wh[id] = (_Float16)v;
}

// ---------------------------------------------------------------------------
// Zero the time halos of an activation buffer for current length T.
// ---------------------------------------------------------------------------
__global__ void halo_zero(_Float16* __restrict__ x, int T)
{
  int id = blockIdx.x * blockDim.x + threadIdx.x;
  if (id >= BATCH * CDIM * 2 * HALO) return;
  int i   = id & (2 * HALO - 1);
  int row = id >> 5;
  int off = (i < HALO) ? i : (T + i);   // [0,16) and [HALO+T, HALO+T+16)
  x[(size_t)row * TSTRIDE + off] = (_Float16)0.f;
}

// ---------------------------------------------------------------------------
// features (B,T,63) f32 -> halo buffer (B, 64ch used, T) f16; channel 63 = 0.
// ---------------------------------------------------------------------------
__global__ void prep_input(const float* __restrict__ f, _Float16* __restrict__ xh)
{
  size_t id = (size_t)blockIdx.x * blockDim.x + threadIdx.x;
  size_t total = (size_t)BATCH * TFULL * 64;
  if (id >= total) return;
  int c = id & 63;
  int t = (id >> 6) % TFULL;
  int b = id / ((size_t)64 * TFULL);
  float v = (c < NFEATS) ? f[((size_t)b * TFULL + t) * NFEATS + c] : 0.f;
  xh[((size_t)b * CDIM + c) * TSTRIDE + HALO + t] = (_Float16)v;
}

// ---------------------------------------------------------------------------
// per-row |x|^2 of xe rows (row = b*TQ + t, over channels), wave per row
// ---------------------------------------------------------------------------
__global__ void rownorm_k(const _Float16* __restrict__ xe, float* __restrict__ xn)
{
  int gw = (blockIdx.x * 256 + threadIdx.x) >> 5;
  int lane = threadIdx.x & 31;
  if (gw >= BATCH * TQ) return;
  int b = gw >> 7, t = gw & (TQ - 1);
  float s = 0.f;
  #pragma unroll
  for (int i = 0; i < WIDTH / 32; ++i) {
    int c = lane + (i << 5);
    float v = (float)xe[((size_t)b * CDIM + c) * TSTRIDE + HALO + t];
    s += v * v;
  }
  #pragma unroll
  for (int mo = 16; mo >= 1; mo >>= 1) s += __shfl_xor(s, mo, 32);
  if (lane == 0) xn[gw] = s;
}

__global__ void cbnorm_k(const float* __restrict__ cb, float* __restrict__ cn)
{
  int gw = (blockIdx.x * 256 + threadIdx.x) >> 5;
  int lane = threadIdx.x & 31;
  if (gw >= CODEN) return;
  float s = 0.f;
  #pragma unroll
  for (int i = 0; i < CODED / 32; ++i) {
    float v = cb[(size_t)gw * CODED + lane + (i << 5)];
    s += v * v;
  }
  #pragma unroll
  for (int mo = 16; mo >= 1; mo >>= 1) s += __shfl_xor(s, mo, 32);
  if (lane == 0) cn[gw] = s;
}

__global__ void init_vq(int* __restrict__ counts, float* __restrict__ accum)
{
  int id = blockIdx.x * blockDim.x + threadIdx.x;
  if (id < CODEN) counts[id] = 0;
  if (id == CODEN) accum[0] = 0.f;
}

// ---------------------------------------------------------------------------
// argmin over codes: dist = |x|^2 - 2*score + |cb|^2 ; scores layout (b,code,t)
// ---------------------------------------------------------------------------
__global__ void argmin_k(const float* __restrict__ scores,
                         const float* __restrict__ xn,
                         const float* __restrict__ cn,
                         int* __restrict__ vidx, int* __restrict__ counts,
                         float* __restrict__ accum)
{
  int gw = (blockIdx.x * 256 + threadIdx.x) >> 5;
  int lane = threadIdx.x & 31;
  if (gw >= BATCH * TQ) return;
  int b = gw >> 7, t = gw & (TQ - 1);
  float x2 = xn[gw];
  float best = 3.4e38f; int bi = 0;
  #pragma unroll
  for (int i = 0; i < CODEN / 32; ++i) {
    int code = lane + (i << 5);
    float d = x2 - 2.f * scores[((size_t)b * CODEN + code) * TQ + t] + cn[code];
    if (d < best) { best = d; bi = code; }
  }
  #pragma unroll
  for (int mo = 16; mo >= 1; mo >>= 1) {
    float ob = __shfl_xor(best, mo, 32);
    int   oi = __shfl_xor(bi, mo, 32);
    if (ob < best || (ob == best && oi < bi)) { best = ob; bi = oi; }
  }
  if (lane == 0) {
    vidx[gw] = bi;
    atomicAdd(&counts[bi], 1);
    atomicAdd(accum, best);   // sum of min dists == sum (xf-xd)^2
  }
}

__global__ void commit_fin(const float* __restrict__ accum, float* __restrict__ o)
{
  o[0] = accum[0] / ((float)(BATCH * TQ) * (float)CODED);
}

__global__ void perp_k(const int* __restrict__ counts, float* __restrict__ o)
{
  __shared__ float sm[CODEN];
  int tid = threadIdx.x;
  float p = (float)counts[tid] * (1.f / (float)(BATCH * TQ));
  sm[tid] = p * logf(p + 1e-10f);
  __syncthreads();
  for (int s = CODEN / 2; s > 0; s >>= 1) {
    if (tid < s) sm[tid] += sm[tid + s];
    __syncthreads();
  }
  if (tid == 0) o[0] = expf(-sm[0]);
}

// ---------------------------------------------------------------------------
// Rotation trick, per (b,c) row of length TQ (norms along T); wave per row.
// ---------------------------------------------------------------------------
__global__ void rot_k(const _Float16* __restrict__ xe,
                      const float* __restrict__ cb,
                      const int* __restrict__ vidx,
                      _Float16* __restrict__ out)
{
  int gw = (blockIdx.x * 256 + threadIdx.x) >> 5;
  int lane = threadIdx.x & 31;
  if (gw >= BATCH * WIDTH) return;
  int b = gw / WIDTH, c = gw % WIDTH;
  const size_t rb = ((size_t)b * CDIM + c) * TSTRIDE + HALO;

  float xev[TQ / 32], xqv[TQ / 32];
  float se = 0.f, sq = 0.f, sc = 0.f;
  #pragma unroll
  for (int i = 0; i < TQ / 32; ++i) {
    int t = lane + (i << 5);
    float e = (float)xe[rb + t];
    float q = cb[(size_t)vidx[b * TQ + t] * CODED + c];
    xev[i] = e; xqv[i] = q;
    se += e * e; sq += q * q; sc += e * q;
  }
  #pragma unroll
  for (int mo = 16; mo >= 1; mo >>= 1) {
    se += __shfl_xor(se, mo, 32);
    sq += __shfl_xor(sq, mo, 32);
    sc += __shfl_xor(sc, mo, 32);
  }
  float ne = fmaxf(sqrtf(se), 1e-12f);
  float nq = fmaxf(sqrtf(sq), 1e-12f);
  float un2 = se / (ne * ne) + 2.f * sc / (ne * nq) + sq / (nq * nq);
  float un = fmaxf(sqrtf(fmaxf(un2, 0.f)), 1e-12f);
  float rx = (se / ne + sc / nq) / un;   // sum(r * xe)
  float ex = se / ne;                    // sum(e_n * xe)
  float scal = nq / ne;
  #pragma unroll
  for (int i = 0; i < TQ / 32; ++i) {
    int t = lane + (i << 5);
    float e = xev[i], q = xqv[i];
    float r = (e / ne + q / nq) / un;
    out[rb + t] = (_Float16)(scal * (e - 2.f * r * rx + 2.f * (q / nq) * ex));
  }
}

// ---------------------------------------------------------------------------
// Linear interp TQ -> TFULL and transpose to (B, T, 63)
// ---------------------------------------------------------------------------
__global__ void interp_k(const float* __restrict__ xd, float* __restrict__ out)
{
  size_t id = (size_t)blockIdx.x * blockDim.x + threadIdx.x;
  size_t total = (size_t)BATCH * TFULL * NFEATS;
  if (id >= total) return;
  int c = id % NFEATS;
  int j = (id / NFEATS) % TFULL;
  int b = id / ((size_t)NFEATS * TFULL);
  float pos = (j + 0.5f) * ((float)TQ / (float)TFULL) - 0.5f;
  pos = fminf(fmaxf(pos, 0.f), (float)(TQ - 1));
  int i0 = (int)pos;
  int i1 = (i0 + 1 < TQ - 1) ? (i0 + 1) : (TQ - 1);
  float w = pos - (float)i0;
  const float* row = xd + ((size_t)b * NFEATS + c) * TQ;
  out[id] = row[i0] * (1.f - w) + row[i1] * w;
}

// ---------------------------------------------------------------------------
// Host driver
// ---------------------------------------------------------------------------
extern "C" void kernel_launch(void* const* d_in, const int* in_sizes, int n_in,
                              void* d_out, int out_size, void* d_ws, size_t ws_size,
                              hipStream_t stream)
{
  (void)in_sizes; (void)n_in; (void)out_size; (void)ws_size;
  const float* features = (const float*)d_in[0];
  auto P = [&](int i) { return (const float*)d_in[i]; };
  const float* cb = (const float*)d_in[95];

  // ---- workspace carve-up
  char* p = (char*)d_ws;
  const size_t HB = (size_t)BATCH * CDIM * TSTRIDE * sizeof(_Float16);
  _Float16* hbuf[3];
  hbuf[0] = (_Float16*)p; p += HB;
  hbuf[1] = (_Float16*)p; p += HB;
  hbuf[2] = (_Float16*)p; p += HB;
  float* scores = (float*)p; p += (size_t)BATCH * CODEN * TQ * sizeof(float);
  float* xnorm  = (float*)p; p += (size_t)BATCH * TQ * sizeof(float);
  float* cnorm  = (float*)p; p += (size_t)CODEN * sizeof(float);
  int*   vidx   = (int*)p;   p += (size_t)BATCH * TQ * sizeof(int);
  int*   counts = (int*)p;   p += (size_t)CODEN * sizeof(int);
  float* accum  = (float*)p; p += 64;
  float* xdecf  = (float*)p; p += (size_t)BATCH * NFEATS * TQ * sizeof(float);
  _Float16* warena = (_Float16*)p;          // packed-weight bump arena

  const int HZGRID = (BATCH * CDIM * 2 * HALO + 255) / 256;

  // one conv layer: pack weights, run WMMA conv, re-zero output halos
  auto CONV = [&](const _Float16* x, const float* wsrc, const float* bp,
                  const _Float16* res, _Float16* yh, float* yf,
                  int Cin, int Cout, int Tin, int K, int stride, int dil,
                  int pad, int relu_in, int relu_out) -> int {
    int Cinp  = (Cin + 31) & ~31;
    int Coutp = (Cout + 127) & ~127;            // 8 M-waves per block
    int Tout  = (Tin + 2 * pad - dil * (K - 1) - 1) / stride + 1;
    int wtot  = Coutp * K * Cinp;
    _Float16* wh = warena; warena += wtot;
    pack_w<<<(wtot + 255) / 256, 256, 0, stream>>>(wsrc, wh, Cout, Cin, K, Cinp, wtot);
    dim3 g(Tout / 64, Coutp / 128, BATCH);
    if (K == 3 && stride == 1 && dil == 1)
      conv1d_wmma<3, 1, 1><<<g, 256, 0, stream>>>(x, wh, bp, res, yh, yf,
          Cinp, Cout, Tout, pad, relu_in, relu_out);
    else if (K == 3 && stride == 1 && dil == 3)
      conv1d_wmma<3, 1, 3><<<g, 256, 0, stream>>>(x, wh, bp, res, yh, yf,
          Cinp, Cout, Tout, pad, relu_in, relu_out);
    else if (K == 3 && stride == 1 && dil == 9)
      conv1d_wmma<3, 1, 9><<<g, 256, 0, stream>>>(x, wh, bp, res, yh, yf,
          Cinp, Cout, Tout, pad, relu_in, relu_out);
    else if (K == 3 && stride == 2)
      conv1d_wmma<3, 2, 1><<<g, 256, 0, stream>>>(x, wh, bp, res, yh, yf,
          Cinp, Cout, Tout, pad, relu_in, relu_out);
    else
      conv1d_wmma<1, 1, 1><<<g, 256, 0, stream>>>(x, wh, bp, res, yh, yf,
          Cinp, Cout, Tout, pad, relu_in, relu_out);
    if (yh) halo_zero<<<HZGRID, 256, 0, stream>>>(yh, Tout);
    return Tout;
  };

  int cur = 0, f1 = 1, f2 = 2;
  auto SWAP = [&]() { int t = cur; cur = f1; f1 = t; };
  auto ROT3 = [&]() { int t = cur; cur = f2; f2 = f1; f1 = t; };

  // ---- input transpose -> f16 halo buffer
  {
    size_t total = (size_t)BATCH * TFULL * 64;
    prep_input<<<(total + 255) / 256, 256, 0, stream>>>(features, hbuf[0]);
    halo_zero<<<HZGRID, 256, 0, stream>>>(hbuf[0], TFULL);
  }
  int T = TFULL;

  // ---- encoder: in conv (63->512, k3, relu out)
  T = CONV(hbuf[cur], P(1), P(2), nullptr, hbuf[f1], nullptr,
           NFEATS, WIDTH, T, 3, 1, 1, 1, 0, 1);
  SWAP();

  // ---- encoder down blocks
  for (int blk = 0; blk < 3; ++blk) {
    int base = 3 + 14 * blk;
    T = CONV(hbuf[cur], P(base), P(base + 1), nullptr, hbuf[f1], nullptr,
             WIDTH, WIDTH, T, 3, 2, 1, 1, 0, 0);
    SWAP();
    int dil = 1;
    for (int j = 0; j < 3; ++j) {
      int wb = base + 2 + 4 * j;
      CONV(hbuf[cur], P(wb), P(wb + 1), nullptr, hbuf[f1], nullptr,
           WIDTH, WIDTH, T, 3, 1, dil, dil, 1, 0);
      CONV(hbuf[f1], P(wb + 2), P(wb + 3), hbuf[cur], hbuf[f2], nullptr,
           WIDTH, WIDTH, T, 1, 1, 1, 0, 1, 0);
      ROT3();
      dil *= 3;
    }
  }

  // ---- encoder out conv (k3)
  T = CONV(hbuf[cur], P(45), P(46), nullptr, hbuf[f1], nullptr,
           WIDTH, WIDTH, T, 3, 1, 1, 1, 0, 0);
  SWAP();
  _Float16* xe = hbuf[cur];   // (B, 512, 128) in halo layout, T == TQ

  // ---- VQ: norms + WMMA codebook GEMM + argmin + stats
  rownorm_k<<<(BATCH * TQ) / 8, 256, 0, stream>>>(xe, xnorm);
  cbnorm_k<<<CODEN / 8, 256, 0, stream>>>(cb, cnorm);
  CONV(xe, cb, nullptr, nullptr, nullptr, scores,
       CODED, CODEN, TQ, 1, 1, 1, 0, 0, 0);
  init_vq<<<3, 256, 0, stream>>>(counts, accum);
  argmin_k<<<(BATCH * TQ) / 8, 256, 0, stream>>>(scores, xnorm, cnorm,
                                                 vidx, counts, accum);
  float* out = (float*)d_out;
  const size_t NOUT = (size_t)BATCH * TFULL * NFEATS;
  commit_fin<<<1, 1, 0, stream>>>(accum, out + NOUT);
  perp_k<<<1, CODEN, 0, stream>>>(counts, out + NOUT + 1);

  // ---- rotation trick: xe -> xq_rot
  rot_k<<<(BATCH * WIDTH) / 8, 256, 0, stream>>>(xe, cb, vidx, hbuf[f1]);
  halo_zero<<<HZGRID, 256, 0, stream>>>(hbuf[f1], TQ);
  SWAP();

  // ---- decoder: in conv (relu out)
  CONV(hbuf[cur], P(47), P(48), nullptr, hbuf[f1], nullptr,
       WIDTH, WIDTH, TQ, 3, 1, 1, 1, 0, 1);
  SWAP();

  // ---- decoder up blocks (dilations reversed: 9,3,1), then k3 conv
  for (int blk = 0; blk < 3; ++blk) {
    int base = 49 + 14 * blk;
    int dil = 9;
    for (int j = 0; j < 3; ++j) {
      int wb = base + 4 * j;
      CONV(hbuf[cur], P(wb), P(wb + 1), nullptr, hbuf[f1], nullptr,
           WIDTH, WIDTH, TQ, 3, 1, dil, dil, 1, 0);
      CONV(hbuf[f1], P(wb + 2), P(wb + 3), hbuf[cur], hbuf[f2], nullptr,
           WIDTH, WIDTH, TQ, 1, 1, 1, 0, 1, 0);
      ROT3();
      dil /= 3;
    }
    CONV(hbuf[cur], P(base + 12), P(base + 13), nullptr, hbuf[f1], nullptr,
         WIDTH, WIDTH, TQ, 3, 1, 1, 1, 0, 0);
    SWAP();
  }

  // ---- decoder mid conv (relu) and out conv (512 -> 63, f32 compact out)
  CONV(hbuf[cur], P(91), P(92), nullptr, hbuf[f1], nullptr,
       WIDTH, WIDTH, TQ, 3, 1, 1, 1, 0, 1);
  SWAP();
  CONV(hbuf[cur], P(93), P(94), nullptr, nullptr, xdecf,
       WIDTH, NFEATS, TQ, 3, 1, 1, 1, 0, 0);

  // ---- interpolate to T=1024 and transpose into d_out (B, T, 63)
  {
    size_t total = (size_t)BATCH * TFULL * NFEATS;
    interp_k<<<(total + 255) / 256, 256, 0, stream>>>(xdecf, out);
  }
}
